// MultiHeadAttention_57913339019722
// MI455X (gfx1250) — matrix operand order repaired and checked by reading, not compiled, not run on hardware
//
#include <hip/hip_runtime.h>

#define B_  4
#define L_  2048
#define S_  2048
#define D_  1024
#define H_  16
#define HD_ 64

typedef __attribute__((ext_vector_type(16))) _Float16 v16h;
typedef __attribute__((ext_vector_type(8)))  _Float16 v8h;
typedef __attribute__((ext_vector_type(8)))  float    v8f;

union V16U { v16h v; v8h h[2]; };

__device__ __forceinline__ v8f wmma_f32_f16(v16h a, v16h b, v8f c) {
  // 8 args: (neg_a, A, neg_b, B, c_mod, C, reuse_a, reuse_b)
  return __builtin_amdgcn_wmma_f32_16x16x32_f16(false, a, false, b, (short)0, c,
                                                false, false);
}

// ---------------------------------------------------------------------------
// GEMM with bias: Out[M,N] = (A[M,K] * W[K,N] + bias[N]) * outScale
// MODE 0: A fp32 -> f16 row-major out             (Q, K projections)
// MODE 1: A fp32 -> f16 transposed out [B,H,HD,S] (V projection)
// MODE 2: A f16  -> fp32 row-major out            (output projection)
// Block tile 128x128, BK=64, 8 waves (2x4 WMMA frags each, 32x64 per wave).
// Double-buffered LDS pipeline: one barrier per K-step.
// The softmax 1/sqrt(HD) is folded into the Q projection via outScale.
// ---------------------------------------------------------------------------
template <int MODE>
__global__ __launch_bounds__(256)
void gemm_bias_kernel(const void* __restrict__ Aptr,
                      const float* __restrict__ W,
                      const float* __restrict__ bias,
                      void* __restrict__ Out,
                      int M, int K, int N, float outScale)
{
  constexpr bool A_F32 = (MODE != 2);
  __shared__ __attribute__((aligned(16))) _Float16 As[2][128][72]; // [m][k]
  __shared__ __attribute__((aligned(16))) _Float16 Bs[2][128][72]; // [n][k]

  const int tid    = threadIdx.x;
  const int lane   = tid & 31;
  const int wave   = tid >> 5;
  const int laneLo = lane & 15;
  const int khalf  = lane >> 4;
  const int waveM  = wave & 3;   // 0..3 -> 32-row strip
  const int waveN  = wave >> 2;  // 0..1 -> 64-col strip
  const int mBase  = blockIdx.x * 128;
  const int nBase  = blockIdx.y * 128;

  const v8f vzero = {0.f, 0.f, 0.f, 0.f, 0.f, 0.f, 0.f, 0.f};
  v8f c[2][4];
  #pragma unroll
  for (int mi = 0; mi < 2; ++mi)
    #pragma unroll
    for (int ni = 0; ni < 4; ++ni)
      c[mi][ni] = vzero;

  float4 aReg[8];
  v8h    aRegH[4];
  float4 wReg[8];

  auto loadGlobal = [&](int k0) {
    if constexpr (A_F32) {
      const float* A = (const float*)Aptr;
      #pragma unroll
      for (int i = 0; i < 8; ++i) {
        const int lin = tid + i * 256;       // 0..2047
        const int m   = lin >> 4;            // 16 float4 per 64-wide row
        const int kk  = (lin & 15) << 2;
        aReg[i] = *(const float4*)(A + (size_t)(mBase + m) * K + k0 + kk);
      }
    } else {
      const _Float16* A = (const _Float16*)Aptr;
      #pragma unroll
      for (int i = 0; i < 4; ++i) {
        const int lin = tid + i * 256;       // 0..1023
        const int m   = lin >> 3;            // 8 v8h per row
        const int kk  = (lin & 7) << 3;
        aRegH[i] = *(const v8h*)(A + (size_t)(mBase + m) * K + k0 + kk);
      }
    }
    #pragma unroll
    for (int i = 0; i < 8; ++i) {
      const int lin = tid + i * 256;         // 0..2047
      const int kq  = lin >> 5;              // 0..63
      const int nn  = (lin & 31) << 2;       // 0..124
      wReg[i] = *(const float4*)(W + (size_t)(k0 + kq) * N + nBase + nn);
    }
  };

  auto storeLds = [&](int buf) {
    if constexpr (A_F32) {
      #pragma unroll
      for (int i = 0; i < 8; ++i) {
        const int lin = tid + i * 256;
        const int m   = lin >> 4;
        const int kk  = (lin & 15) << 2;
        As[buf][m][kk + 0] = (_Float16)aReg[i].x;
        As[buf][m][kk + 1] = (_Float16)aReg[i].y;
        As[buf][m][kk + 2] = (_Float16)aReg[i].z;
        As[buf][m][kk + 3] = (_Float16)aReg[i].w;
      }
    } else {
      #pragma unroll
      for (int i = 0; i < 4; ++i) {
        const int lin = tid + i * 256;
        const int m   = lin >> 3;
        const int kk  = (lin & 7) << 3;
        *(v8h*)&As[buf][m][kk] = aRegH[i];
      }
    }
    #pragma unroll
    for (int i = 0; i < 8; ++i) {
      const int lin = tid + i * 256;
      const int kq  = lin >> 5;
      const int nn  = (lin & 31) << 2;
      Bs[buf][nn + 0][kq] = (_Float16)wReg[i].x;
      Bs[buf][nn + 1][kq] = (_Float16)wReg[i].y;
      Bs[buf][nn + 2][kq] = (_Float16)wReg[i].z;
      Bs[buf][nn + 3][kq] = (_Float16)wReg[i].w;
    }
  };

  auto compute = [&](int buf) {
    #pragma unroll
    for (int ks = 0; ks < 2; ++ks) {
      V16U a[2], b[4];
      #pragma unroll
      for (int mi = 0; mi < 2; ++mi) {
        // CDNA5 16-bit A 16x32: lane%16 = row, two 8-half K runs
        const _Float16* ap =
            &As[buf][waveM * 32 + mi * 16 + laneLo][ks * 32 + khalf * 8];
        a[mi].h[0] = *(const v8h*)ap;
        a[mi].h[1] = *(const v8h*)(ap + 16);
      }
      #pragma unroll
      for (int ni = 0; ni < 4; ++ni) {
        // B 32x16: lane%16 = col, two 8-half K runs (same order as A)
        const _Float16* bp =
            &Bs[buf][waveN * 64 + ni * 16 + laneLo][ks * 32 + khalf * 8];
        b[ni].h[0] = *(const v8h*)bp;
        b[ni].h[1] = *(const v8h*)(bp + 16);
      }
      #pragma unroll
      for (int mi = 0; mi < 2; ++mi)
        #pragma unroll
        for (int ni = 0; ni < 4; ++ni)
          c[mi][ni] = wmma_f32_f16(a[mi].v, b[ni].v, c[mi][ni]);
    }
  };

  // software-pipelined K loop
  const int KT = K / 64;
  loadGlobal(0);
  storeLds(0);
  __syncthreads();
  for (int kt = 0; kt < KT; ++kt) {
    const int cur = kt & 1;
    if (kt + 1 < KT) loadGlobal((kt + 1) * 64);
    compute(cur);
    if (kt + 1 < KT) storeLds(cur ^ 1);
    __syncthreads();
  }

  // ---- epilogue ----
  #pragma unroll
  for (int ni = 0; ni < 4; ++ni) {
    const int gc = nBase + waveN * 64 + ni * 16 + laneLo;
    const float bv = bias[gc];
    #pragma unroll
    for (int mi = 0; mi < 2; ++mi) {
      const int gr0 = mBase + waveM * 32 + mi * 16 + khalf * 8;
      if constexpr (MODE == 0) {
        #pragma unroll
        for (int r = 0; r < 8; ++r)
          ((_Float16*)Out)[(size_t)(gr0 + r) * N + gc] =
              (_Float16)((c[mi][ni][r] + bv) * outScale);
      } else if constexpr (MODE == 1) {
        // transposed V: [B,H,HD,S]; C-frag rows are consecutive tokens
        const int bIdx = mBase / L_;              // 128-row tile never crosses batch
        const int sTok = gr0 - bIdx * L_;
        const int hh = gc >> 6;
        const int dd = gc & 63;
        v8h pack;
        #pragma unroll
        for (int r = 0; r < 8; ++r)
          pack[r] = (_Float16)(c[mi][ni][r] + bv);
        *(v8h*)(((_Float16*)Out) +
                (((size_t)bIdx * H_ + hh) * HD_ + dd) * S_ + sTok) = pack;
      } else {
        #pragma unroll
        for (int r = 0; r < 8; ++r)
          ((float*)Out)[(size_t)(gr0 + r) * N + gc] = c[mi][ni][r] + bv;
      }
    }
  }
}

// ---------------------------------------------------------------------------
// Flash attention: per (b, h, 128 Q rows). 8 independent waves x 16 rows.
// Q is pre-scaled by 1/sqrt(HD) at projection time, so scores come straight
// out of the WMMA ready for masking/softmax. V arrives pre-transposed
// [B,H,HD,S] so P.V B-frags are contiguous global loads -> zero workgroup
// barriers; each wave streams S in 64-col chunks up to its own causal
// frontier. Row-sum kept as per-lane partials (one reduction at the end);
// only the row-max needs a per-chunk 16-lane reduction.
// ---------------------------------------------------------------------------
__global__ __launch_bounds__(256)
void attn_kernel(const _Float16* __restrict__ Qh,
                 const _Float16* __restrict__ Kh,
                 const _Float16* __restrict__ VhT,
                 _Float16* __restrict__ Att)
{
  __shared__ __attribute__((aligned(16))) _Float16 Pl[8][16][72]; // per-wave P

  const int tid    = threadIdx.x;
  const int lane   = tid & 31;
  const int wave   = tid >> 5;
  const int laneLo = lane & 15;
  const int khalf  = lane >> 4;

  const int qBase = blockIdx.x * 128;
  const int b     = blockIdx.y / H_;
  const int h     = blockIdx.y % H_;

  const _Float16* Qb = Qh + (size_t)b * L_ * D_ + h * HD_;
  const _Float16* Kb = Kh + (size_t)b * S_ * D_ + h * HD_;
  const _Float16* Vb = VhT + ((size_t)b * H_ + h) * HD_ * S_;   // [d][s]

  // Q fragments (A layout: lane%16 = row)
  const int rowQ = qBase + wave * 16 + laneLo;
  V16U qa[2];
  #pragma unroll
  for (int ks = 0; ks < 2; ++ks) {
    const _Float16* qp = Qb + (size_t)rowQ * D_ + ks * 32 + khalf * 8;
    qa[ks].h[0] = *(const v8h*)qp;
    qa[ks].h[1] = *(const v8h*)(qp + 16);
  }

  const v8f vzero = {0.f, 0.f, 0.f, 0.f, 0.f, 0.f, 0.f, 0.f};
  v8f o[4] = {vzero, vzero, vzero, vzero};
  float mrow[8], lpart[8];
  #pragma unroll
  for (int r = 0; r < 8; ++r) { mrow[r] = -1e30f; lpart[r] = 0.0f; }

  const int sEndW = qBase + wave * 16 + 16;    // this wave's causal frontier

  for (int s0 = 0; s0 < sEndW; s0 += 64) {
    if (s0 + 64 < sEndW) {  // hint next chunk (-> global_prefetch_b8)
      __builtin_prefetch((const void*)(Kb + (size_t)(s0 + 64 + lane) * D_), 0, 2);
      __builtin_prefetch((const void*)(Vb + (size_t)(lane << 1) * S_ + s0 + 64), 0, 2);
    }

    // ---- scores = (Q/sqrt(HD)) K^T over 64 columns (4 frags x 2 K-steps) ----
    v8f sc[4];
    #pragma unroll
    for (int nb = 0; nb < 4; ++nb) {
      sc[nb] = vzero;
      #pragma unroll
      for (int ks = 0; ks < 2; ++ks) {
        V16U kf;
        const _Float16* kp =
            Kb + (size_t)(s0 + nb * 16 + laneLo) * D_ + ks * 32 + khalf * 8;
        kf.h[0] = *(const v8h*)kp;
        kf.h[1] = *(const v8h*)(kp + 16);
        sc[nb] = wmma_f32_f16(qa[ks].v, kf.v, sc[nb]);
      }
    }

    // ---- online softmax in C-fragment layout (scores already scaled) ----
    const int rowBase = qBase + wave * 16 + khalf * 8;
    #pragma unroll
    for (int r = 0; r < 8; ++r) {
      const int grow = rowBase + r;
      float p[4];
      float mx = -1e30f;
      #pragma unroll
      for (int nb = 0; nb < 4; ++nb) {
        float s = sc[nb][r];
        if (s0 + nb * 16 + laneLo > grow) s = -1e30f;   // causal mask
        p[nb] = s;
        mx = fmaxf(mx, s);
      }
      #pragma unroll
      for (int off = 1; off < 16; off <<= 1)
        mx = fmaxf(mx, __shfl_xor(mx, off, 32));
      const float mnew  = fmaxf(mrow[r], mx);
      const float alpha = __expf(mrow[r] - mnew);
      float psum = 0.f;
      #pragma unroll
      for (int nb = 0; nb < 4; ++nb) {
        p[nb] = __expf(p[nb] - mnew);
        psum += p[nb];
      }
      lpart[r] = lpart[r] * alpha + psum;   // per-lane partial denominator
      mrow[r]  = mnew;
      #pragma unroll
      for (int nb = 0; nb < 4; ++nb)
        o[nb][r] = o[nb][r] * alpha;
      #pragma unroll
      for (int nb = 0; nb < 4; ++nb)
        Pl[wave][khalf * 8 + r][nb * 16 + laneLo] = (_Float16)p[nb];
    }
    // wave-private LDS bounce; DS in-order per wave, make the wait explicit
    __asm__ volatile("s_wait_dscnt 0" ::: "memory");

    // ---- O += P V  (P: 16x64 -> 2 A-frags; V: 64x64 -> 8 B-frags) ----
    #pragma unroll
    for (int ks2 = 0; ks2 < 2; ++ks2) {
      V16U pa;
      const _Float16* pp = &Pl[wave][laneLo][ks2 * 32 + khalf * 8];
      pa.h[0] = *(const v8h*)pp;
      pa.h[1] = *(const v8h*)(pp + 16);
      #pragma unroll
      for (int nb = 0; nb < 4; ++nb) {
        V16U vf;
        const _Float16* vp =
            Vb + (size_t)(nb * 16 + laneLo) * S_ + s0 + ks2 * 32 + khalf * 8;
        vf.h[0] = *(const v8h*)vp;
        vf.h[1] = *(const v8h*)(vp + 16);
        o[nb] = wmma_f32_f16(pa.v, vf.v, o[nb]);
      }
    }
  }

  // ---- final: one row-sum reduction, normalize, store f16 ----
  #pragma unroll
  for (int r = 0; r < 8; ++r) {
    float ls = lpart[r];
    #pragma unroll
    for (int off = 1; off < 16; off <<= 1)
      ls += __shfl_xor(ls, off, 32);
    const float inv = 1.0f / ls;
    const int grow = qBase + wave * 16 + khalf * 8 + r;
    _Float16* op = Att + ((size_t)b * L_ + grow) * D_ + h * HD_;
    #pragma unroll
    for (int nb = 0; nb < 4; ++nb)
      op[nb * 16 + laneLo] = (_Float16)(o[nb][r] * inv);
  }
}

// ---------------------------------------------------------------------------
extern "C" void kernel_launch(void* const* d_in, const int* in_sizes, int n_in,
                              void* d_out, int out_size, void* d_ws, size_t ws_size,
                              hipStream_t stream) {
  (void)in_sizes; (void)n_in; (void)out_size; (void)ws_size;
  const float* q  = (const float*)d_in[0];
  const float* k  = (const float*)d_in[1];
  const float* v  = (const float*)d_in[2];
  const float* Wq = (const float*)d_in[3];
  const float* bq = (const float*)d_in[4];
  const float* Wk = (const float*)d_in[5];
  const float* bk = (const float*)d_in[6];
  const float* Wv = (const float*)d_in[7];
  const float* bv = (const float*)d_in[8];
  const float* Wo = (const float*)d_in[9];
  const float* bo = (const float*)d_in[10];

  const size_t nTok = (size_t)B_ * L_ * D_;   // 8,388,608 elements
  _Float16* Qh  = (_Float16*)d_ws;            // 16 MB each, 64 MB total
  _Float16* Kh  = Qh + nTok;
  _Float16* VhT = Kh + nTok;                  // transposed [B,H,HD,S]
  _Float16* At  = VhT + nTok;

  const int M = B_ * L_;   // 8192
  const float qScale = 0.125f;   // 1/sqrt(HD) folded into Q projection
  dim3 blk(256);
  dim3 gGemm(M / 128, D_ / 128);   // (64, 8)

  gemm_bias_kernel<0><<<gGemm, blk, 0, stream>>>(q, Wq, bq, Qh,  M, D_, D_, qScale);
  gemm_bias_kernel<0><<<gGemm, blk, 0, stream>>>(k, Wk, bk, Kh,  M, D_, D_, 1.0f);
  gemm_bias_kernel<1><<<gGemm, blk, 0, stream>>>(v, Wv, bv, VhT, M, D_, D_, 1.0f);

  dim3 gAttn(L_ / 128, B_ * H_);   // (16, 64)
  attn_kernel<<<gAttn, blk, 0, stream>>>(Qh, Kh, VhT, At);

  gemm_bias_kernel<2><<<gGemm, blk, 0, stream>>>(At, Wo, bo, d_out, M, D_, D_, 1.0f);
}
